// ResBlock_3813930959554
// MI455X (gfx1250) — matrix operand order repaired and checked by reading, not compiled, not run on hardware
//
#include <hip/hip_runtime.h>
#include <math.h>

// ---------------------------------------------------------------------------
// ResBlock: out = x + gelu_exact( LayerNorm(x) @ W^T + b )
// x: (M=16384, D=1024) f32, W: (1024,1024) f32 (torch Linear [out,in])
//
// MI455X / gfx1250 (wave32, WMMA):
//   pass 1: LayerNorm rows of x -> h (bf16) workspace
//   pass 2: W -> bf16 workspace (row-major; WMMA B-fragment = contiguous W rows)
//   pass 3: 128x128-tile GEMM, V_WMMA_F32_16X16X32_BF16 (f32 accum),
//           double-buffered LDS staged with GLOBAL_LOAD_ASYNC_TO_LDS_B128
//           (ASYNCcnt) when available, fused bias + exact GELU + residual.
// ---------------------------------------------------------------------------

typedef __attribute__((ext_vector_type(16))) __bf16 v16b;
typedef __attribute__((ext_vector_type(8)))  __bf16 v8b;
typedef __attribute__((ext_vector_type(8)))  float  v8f;

#define DD 1024

static __device__ __forceinline__ unsigned short f2bf(float f) {
  // round-to-nearest-even f32 -> bf16 (inputs are finite)
  unsigned int u = __builtin_bit_cast(unsigned int, f);
  u += 0x7fffu + ((u >> 16) & 1u);
  return (unsigned short)(u >> 16);
}

// -------------------- pass 1: LayerNorm -> bf16 h --------------------------
__global__ __launch_bounds__(256) void ln_kernel(const float* __restrict__ x,
                                                 const float* __restrict__ gamma,
                                                 const float* __restrict__ beta,
                                                 unsigned short* __restrict__ h) {
  __shared__ float2 red[256];
  const int row = blockIdx.x;
  const int t = threadIdx.x;

  const float4 xv = ((const float4*)(x + (size_t)row * DD))[t];
  float s = xv.x + xv.y + xv.z + xv.w;
  float q = xv.x * xv.x + xv.y * xv.y + xv.z * xv.z + xv.w * xv.w;
  red[t] = make_float2(s, q);
  __syncthreads();
#pragma unroll
  for (int off = 128; off > 0; off >>= 1) {
    if (t < off) {
      float2 o = red[t + off];
      float2 m = red[t];
      red[t] = make_float2(m.x + o.x, m.y + o.y);
    }
    __syncthreads();
  }
  const float mean = red[0].x * (1.0f / DD);
  const float var  = red[0].y * (1.0f / DD) - mean * mean;
  const float rstd = rsqrtf(var + 1e-5f);

  const int c = t * 4;
  float vals[4] = {xv.x, xv.y, xv.z, xv.w};
  unsigned short r[4];
#pragma unroll
  for (int i = 0; i < 4; ++i) {
    float hn = (vals[i] - mean) * rstd * gamma[c + i] + beta[c + i];
    r[i] = f2bf(hn);
  }
  uint2 packed;
  packed.x = (unsigned int)r[0] | ((unsigned int)r[1] << 16);
  packed.y = (unsigned int)r[2] | ((unsigned int)r[3] << 16);
  ((uint2*)(h + (size_t)row * DD))[t] = packed;
}

// -------------------- pass 2: W (f32) -> bf16 ------------------------------
__global__ __launch_bounds__(256) void wconv_kernel(const float* __restrict__ W,
                                                    unsigned short* __restrict__ Wb) {
  const size_t i = (size_t)blockIdx.x * blockDim.x + threadIdx.x; // 4 elems each
  const float4 w4 = ((const float4*)W)[i];
  uint2 p;
  p.x = (unsigned int)f2bf(w4.x) | ((unsigned int)f2bf(w4.y) << 16);
  p.y = (unsigned int)f2bf(w4.z) | ((unsigned int)f2bf(w4.w) << 16);
  ((uint2*)Wb)[i] = p;
}

// -------------------- pass 3: WMMA GEMM + epilogue -------------------------
#define LDS_STRIDE 40  // 32 + 8 pad: 80B pitch -> conflict-free 16-row frag reads

#if __has_builtin(__builtin_amdgcn_global_load_async_to_lds_b128) && \
    __has_builtin(__builtin_amdgcn_s_wait_asynccnt)
#define USE_ASYNC_LDS 1
// Toolchain signature (from hipcc diagnostic):
//   param0: __attribute__((vector_size(16))) int __device__*   (AS1, non-const)
//   param1: LDS pointer, same vector pointee (AS3)
typedef int v4i_ __attribute__((vector_size(16)));
typedef __attribute__((address_space(1))) v4i_* gv4p;
typedef __attribute__((address_space(3))) v4i_* lv4p;
static __device__ __forceinline__ void async_cp16(const void* g, void* l) {
  __builtin_amdgcn_global_load_async_to_lds_b128((gv4p)g, (lv4p)l, 0, 0);
}
#else
#define USE_ASYNC_LDS 0
#endif

static __device__ __forceinline__ v16b mk16(const unsigned short* p0,
                                            const unsigned short* p1) {
  v8b lo = *(const v8b*)p0;
  v8b hi = *(const v8b*)p1;
  v16b r;
#pragma unroll
  for (int i = 0; i < 8; ++i) { r[i] = lo[i]; r[i + 8] = hi[i]; }
  return r;
}

// 8 WMMAs on one staged 128x32 A-tile / 128x32 B-tile
static __device__ __forceinline__ void compute_tile(const unsigned short* As,
                                                    const unsigned short* Bs,
                                                    int wm, int wn, int lr, int kh,
                                                    v8f (&acc)[2][4]) {
  // A frag: lane holds row (lane&15); VGPR0-3 = K[kh*8..+7], VGPR4-7 = K[kh*8+16..+23]
  v16b afrag[2];
#pragma unroll
  for (int m = 0; m < 2; ++m) {
    const unsigned short* ap = &As[(wm + m * 16 + lr) * LDS_STRIDE + kh * 8];
    afrag[m] = mk16(ap, ap + 16);
  }
  // B frag: lane holds column (lane&15); VGPR0-3 = K[kh*8..+7], VGPR4-7 = K[kh*8+16..+23]
  v16b bfrag[4];
#pragma unroll
  for (int n = 0; n < 4; ++n) {
    const unsigned short* bp = &Bs[(wn + n * 16 + lr) * LDS_STRIDE + kh * 8];
    bfrag[n] = mk16(bp, bp + 16);
  }
#pragma unroll
  for (int m = 0; m < 2; ++m)
#pragma unroll
    for (int n = 0; n < 4; ++n)
      acc[m][n] = __builtin_amdgcn_wmma_f32_16x16x32_bf16(
          false, afrag[m], false, bfrag[n], (short)0, acc[m][n], false, false);
}

__global__ __launch_bounds__(256) void gemm_kernel(const unsigned short* __restrict__ h,
                                                   const unsigned short* __restrict__ Wb,
                                                   const float* __restrict__ bias,
                                                   const float* __restrict__ x,
                                                   float* __restrict__ out) {
  const int t     = threadIdx.x;
  const int tileN = blockIdx.x * 128;
  const int tileM = blockIdx.y * 128;
  const int lane  = t & 31;
  const int wave  = t >> 5;
  const int wm    = (wave & 3) * 32;  // wave M origin in tile
  const int wn    = (wave >> 2) * 64; // wave N origin in tile
  const int lr    = lane & 15;
  const int kh    = lane >> 4;

  const int ldRow = t >> 2;       // 0..63 (two passes cover 128 rows)
  const int ldCol = (t & 3) * 8;  // 8 bf16 (16B) chunk within K=32

  v8f acc[2][4];
#pragma unroll
  for (int m = 0; m < 2; ++m)
#pragma unroll
    for (int n = 0; n < 4; ++n) acc[m][n] = v8f{};

#if USE_ASYNC_LDS
  // ---- double-buffered LDS, staged by GLOBAL_LOAD_ASYNC_TO_LDS_B128 ----
  __shared__ __align__(16) unsigned short As[2][128 * LDS_STRIDE];
  __shared__ __align__(16) unsigned short Bs[2][128 * LDS_STRIDE];

  // prologue: tile k0=0 -> buffer 0 (4 async ops per thread)
#pragma unroll
  for (int i = 0; i < 2; ++i) {
    async_cp16(h  + (size_t)(tileM + ldRow + i * 64) * DD + ldCol,
               &As[0][(ldRow + i * 64) * LDS_STRIDE + ldCol]);
    async_cp16(Wb + (size_t)(tileN + ldRow + i * 64) * DD + ldCol,
               &Bs[0][(ldRow + i * 64) * LDS_STRIDE + ldCol]);
  }

  for (int k0 = 0; k0 < DD; k0 += 32) {
    const int cur = (k0 >> 5) & 1;
    const bool more = (k0 + 32) < DD;
    if (more) { // start next tile into the other buffer
#pragma unroll
      for (int i = 0; i < 2; ++i) {
        async_cp16(h  + (size_t)(tileM + ldRow + i * 64) * DD + (k0 + 32) + ldCol,
                   &As[cur ^ 1][(ldRow + i * 64) * LDS_STRIDE + ldCol]);
        async_cp16(Wb + (size_t)(tileN + ldRow + i * 64) * DD + (k0 + 32) + ldCol,
                   &Bs[cur ^ 1][(ldRow + i * 64) * LDS_STRIDE + ldCol]);
      }
      __builtin_amdgcn_s_wait_asynccnt(4); // async loads complete in order:
                                           // current tile's 4 ops are done
    } else {
      __builtin_amdgcn_s_wait_asynccnt(0);
    }
    __syncthreads(); // publish buffer `cur` across the workgroup
    compute_tile(&As[cur][0], &Bs[cur][0], wm, wn, lr, kh, acc);
    __syncthreads(); // all reads of `cur` done before it is refilled next iter
  }
#else
  // ---- fallback: single LDS buffer, register-staged prefetch ----
  __shared__ __align__(16) unsigned short As1[128 * LDS_STRIDE];
  __shared__ __align__(16) unsigned short Bs1[128 * LDS_STRIDE];

  uint4 aReg[2], bReg[2];
#pragma unroll
  for (int i = 0; i < 2; ++i) {
    aReg[i] = *(const uint4*)(h  + (size_t)(tileM + ldRow + i * 64) * DD + ldCol);
    bReg[i] = *(const uint4*)(Wb + (size_t)(tileN + ldRow + i * 64) * DD + ldCol);
  }
#pragma unroll
  for (int i = 0; i < 2; ++i) {
    *(uint4*)(&As1[(ldRow + i * 64) * LDS_STRIDE + ldCol]) = aReg[i];
    *(uint4*)(&Bs1[(ldRow + i * 64) * LDS_STRIDE + ldCol]) = bReg[i];
  }
  for (int k0 = 0; k0 < DD; k0 += 32) {
    __syncthreads();
    const bool more = (k0 + 32) < DD;
    if (more) {
#pragma unroll
      for (int i = 0; i < 2; ++i) {
        aReg[i] = *(const uint4*)(h  + (size_t)(tileM + ldRow + i * 64) * DD + (k0 + 32) + ldCol);
        bReg[i] = *(const uint4*)(Wb + (size_t)(tileN + ldRow + i * 64) * DD + (k0 + 32) + ldCol);
      }
    }
    compute_tile(As1, Bs1, wm, wn, lr, kh, acc);
    __syncthreads();
    if (more) {
#pragma unroll
      for (int i = 0; i < 2; ++i) {
        *(uint4*)(&As1[(ldRow + i * 64) * LDS_STRIDE + ldCol]) = aReg[i];
        *(uint4*)(&Bs1[(ldRow + i * 64) * LDS_STRIDE + ldCol]) = bReg[i];
      }
    }
  }
#endif

  // epilogue: bias + exact GELU + residual. C/D layout: VGPR r -> M = r + 8*kh, N = lane&15
#pragma unroll
  for (int m = 0; m < 2; ++m) {
#pragma unroll
    for (int n = 0; n < 4; ++n) {
      const int col = tileN + wn + n * 16 + lr;
      const float bv = bias[col];
      const int rowBase = tileM + wm + m * 16 + kh * 8;
#pragma unroll
      for (int r = 0; r < 8; ++r) {
        const size_t idx = (size_t)(rowBase + r) * DD + col;
        const float y = acc[m][n][r] + bv;
        const float g = 0.5f * y * (1.0f + erff(y * 0.70710678118654752f));
        out[idx] = x[idx] + g;
      }
    }
  }
}

// ---------------------------------------------------------------------------
extern "C" void kernel_launch(void* const* d_in, const int* in_sizes, int n_in,
                              void* d_out, int out_size, void* d_ws, size_t ws_size,
                              hipStream_t stream) {
  const float* x     = (const float*)d_in[0];
  const float* W     = (const float*)d_in[1];
  const float* b     = (const float*)d_in[2];
  const float* gamma = (const float*)d_in[3];
  const float* beta  = (const float*)d_in[4];
  float* out = (float*)d_out;

  const int M = in_sizes[0] / DD; // 16384

  unsigned short* h  = (unsigned short*)d_ws;                               // M*D bf16
  unsigned short* Wb = (unsigned short*)((char*)d_ws + (size_t)M * DD * 2); // D*D bf16

  ln_kernel<<<M, 256, 0, stream>>>(x, gamma, beta, h);
  wconv_kernel<<<(DD * DD / 4) / 256, 256, 0, stream>>>(W, Wb);
  dim3 grid(DD / 128, M / 128);
  gemm_kernel<<<grid, 256, 0, stream>>>(h, Wb, b, x, out);
}
